// AttentionTypes_35966056137547
// MI455X (gfx1250) — matrix operand order repaired and checked by reading, not compiled, not run on hardware
//
#include <hip/hip_runtime.h>

#define TPB 256

typedef __bf16 bf16;
typedef __attribute__((ext_vector_type(16))) __bf16 bf16x16;
typedef __attribute__((ext_vector_type(8)))  __bf16 bf16x8;
typedef __attribute__((ext_vector_type(8)))  float   f32x8;

constexpr int B_  = 4, N_ = 2048, M_ = 256, D_ = 1024, CD_ = 768;
constexpr int H_  = 8, DH_ = 64, HD_ = H_ * DH_;       // 512
constexpr int J_  = M_ + 1 + N_;                       // 2305
constexpr int JP_ = 2336;                              // J padded to multiple of 32
constexpr int ROWS_  = B_ * N_;                        // 8192
constexpr int CROWS_ = B_ * M_;                        // 1024

// ---------------------------------------------------------------- WMMA helpers

static __device__ __forceinline__ f32x8 wmma_bf16(bf16x16 a, bf16x16 b, f32x8 c) {
  // v_wmma_f32_16x16x32_bf16  D = A(16x32) * B(32x16) + C
  return __builtin_amdgcn_wmma_f32_16x16x32_bf16(false, a, false, b, (short)0, c,
                                                 false, false);
}

// A operand (16x32 bf16, row striped in-lane):
//   lane row = r0 + (lane&15); K = k0 + 16*half + {0..15}
static __device__ __forceinline__ bf16x16 load_a_op(const bf16* base, int ld, int r0,
                                                    int k0, int lane) {
  const bf16* p = base + (size_t)(r0 + (lane & 15)) * ld + k0 + ((lane >> 4) << 4);
  bf16x8 lo = *(const bf16x8*)p;
  bf16x8 hi = *(const bf16x8*)(p + 8);
  return __builtin_shufflevector(lo, hi, 0, 1, 2, 3, 4, 5, 6, 7,
                                 8, 9, 10, 11, 12, 13, 14, 15);
}

// B operand (32x16 bf16, row striped across lanes): source stored "col-major as rows"
//   lane col = c0 + (lane&15); K = k0 + 16*half + {0..15} (contiguous 32B)
static __device__ __forceinline__ bf16x16 load_b_op(const bf16* base, int ld, int c0,
                                                    int k0, int lane) {
  const bf16* p = base + (size_t)(c0 + (lane & 15)) * ld + k0 + ((lane >> 4) << 4);
  return *(const bf16x16*)p;
}

// ---------------------------------------------------------------- LayerNorm

__global__ void ln_rows_bf16(const float* __restrict__ x, const float* __restrict__ g,
                             const float* __restrict__ bb, bf16* __restrict__ out, int L) {
  int row = blockIdx.x;
  const float* xr = x + (size_t)row * L;
  float s = 0.f, s2 = 0.f;
  for (int i = threadIdx.x; i < L; i += TPB) { float v = xr[i]; s += v; s2 += v * v; }
  __shared__ float rs[8], rs2[8];
  #pragma unroll
  for (int off = 16; off; off >>= 1) { s += __shfl_xor(s, off); s2 += __shfl_xor(s2, off); }
  if ((threadIdx.x & 31) == 0) { rs[threadIdx.x >> 5] = s; rs2[threadIdx.x >> 5] = s2; }
  __syncthreads();
  s = 0.f; s2 = 0.f;
  #pragma unroll
  for (int w = 0; w < 8; w++) { s += rs[w]; s2 += rs2[w]; }
  float mu  = s / L;
  float inv = rsqrtf(fmaxf(s2 / L - mu * mu, 0.f) + 1e-5f);
  bf16* orow = out + (size_t)row * L;
  for (int i = threadIdx.x; i < L; i += TPB)
    orow[i] = (bf16)((xr[i] - mu) * inv * g[i] + bb[i]);
}

// final LN: NT stores (output written once, never re-read on device)
__global__ void ln_rows_f32(const float* __restrict__ x, const float* __restrict__ g,
                            const float* __restrict__ bb, float* __restrict__ out, int L) {
  int row = blockIdx.x;
  const float* xr = x + (size_t)row * L;
  float s = 0.f, s2 = 0.f;
  for (int i = threadIdx.x; i < L; i += TPB) { float v = xr[i]; s += v; s2 += v * v; }
  __shared__ float rs[8], rs2[8];
  #pragma unroll
  for (int off = 16; off; off >>= 1) { s += __shfl_xor(s, off); s2 += __shfl_xor(s2, off); }
  if ((threadIdx.x & 31) == 0) { rs[threadIdx.x >> 5] = s; rs2[threadIdx.x >> 5] = s2; }
  __syncthreads();
  s = 0.f; s2 = 0.f;
  #pragma unroll
  for (int w = 0; w < 8; w++) { s += rs[w]; s2 += rs2[w]; }
  float mu  = s / L;
  float inv = rsqrtf(fmaxf(s2 / L - mu * mu, 0.f) + 1e-5f);
  float* orow = out + (size_t)row * L;
  for (int i = threadIdx.x; i < L; i += TPB)
    __builtin_nontemporal_store((xr[i] - mu) * inv * g[i] + bb[i], orow + i);
}

// ------------------------------------------------- weight transpose fp32->bf16

__global__ void transpose_bf16(const float* __restrict__ src, bf16* __restrict__ dst,
                               int K, int N) {
  int idx = blockIdx.x * TPB + threadIdx.x;
  if (idx >= K * N) return;
  int k = idx / N, n = idx - k * N;
  dst[(size_t)n * K + k] = (bf16)src[idx];
}

// ---------------------------------------------------------------- projections
// All GEMMs computed transposed: C^T(feature, row) = W^T * X^T so that both
// operands are contiguous loads and stores are contiguous vector stores.

__global__ void gemm_q_kernel(const bf16* __restrict__ Wqt, const bf16* __restrict__ xn,
                              bf16* __restrict__ qout) {
  int lane = threadIdx.x & 31;
  int tile = blockIdx.x * 8 + (threadIdx.x >> 5);
  int c0 = (tile & 31) << 4;   // 512/16 feature tiles
  int n0 = (tile >> 5) << 4;   // 8192/16 row tiles
  f32x8 acc = {};
  for (int k0 = 0; k0 < D_; k0 += 32)
    acc = wmma_bf16(load_a_op(Wqt, D_, c0, k0, lane),
                    load_b_op(xn, D_, n0, k0, lane), acc);
  int n = n0 + (lane & 15);
  bf16x8 o;
  #pragma unroll
  for (int r = 0; r < 8; r++) o[r] = (bf16)(acc[r] * 0.125f);  // * DH^-0.5
  *(bf16x8*)(qout + (size_t)n * HD_ + c0 + ((lane >> 4) << 3)) = o;
}

__global__ void gemm_kv_kernel(const bf16* __restrict__ Wkvt, const bf16* __restrict__ xn,
                               bf16* __restrict__ Kbuf, bf16* __restrict__ Vt) {
  int lane = threadIdx.x & 31;
  int tile = blockIdx.x * 8 + (threadIdx.x >> 5);
  int c0 = (tile & 7) << 4;    // 128/16 feature tiles
  int n0 = (tile >> 3) << 4;
  f32x8 acc = {};
  for (int k0 = 0; k0 < D_; k0 += 32)
    acc = wmma_bf16(load_a_op(Wkvt, D_, c0, k0, lane),
                    load_b_op(xn, D_, n0, k0, lane), acc);
  int n = n0 + (lane & 15);
  int b = n >> 11, j = 257 + (n & (N_ - 1));   // self kv lives at j = 257..2304
  int half = lane >> 4;
  if (c0 < DH_) {              // K rows (row-major)
    bf16x8 o;
    #pragma unroll
    for (int r = 0; r < 8; r++) o[r] = (bf16)acc[r];
    *(bf16x8*)(Kbuf + ((size_t)b * JP_ + j) * DH_ + c0 + (half << 3)) = o;
  } else {                     // V stored transposed (DH x Jp)
    int d0 = c0 - DH_ + (half << 3);
    #pragma unroll
    for (int r = 0; r < 8; r++)
      Vt[((size_t)b * DH_ + d0 + r) * JP_ + j] = (bf16)acc[r];
  }
}

__global__ void gemm_ctx_kernel(const bf16* __restrict__ Wct, const bf16* __restrict__ cn,
                                const float* __restrict__ bc, bf16* __restrict__ Kbuf,
                                bf16* __restrict__ Vt) {
  int lane = threadIdx.x & 31;
  int tile = blockIdx.x * 8 + (threadIdx.x >> 5);
  int c0 = (tile & 7) << 4;
  int m0 = (tile >> 3) << 4;
  f32x8 acc = {};
  for (int k0 = 0; k0 < CD_; k0 += 32)
    acc = wmma_bf16(load_a_op(Wct, CD_, c0, k0, lane),
                    load_b_op(cn, CD_, m0, k0, lane), acc);
  int row = m0 + (lane & 15);
  int b = row >> 8, j = row & (M_ - 1);        // context kv at j = 0..255
  int half = lane >> 4;
  int cb = c0 + (half << 3);
  if (c0 < DH_) {
    bf16x8 o;
    #pragma unroll
    for (int r = 0; r < 8; r++) o[r] = (bf16)(acc[r] + bc[cb + r]);
    *(bf16x8*)(Kbuf + ((size_t)b * JP_ + j) * DH_ + c0 + (half << 3)) = o;
  } else {
    int d0 = cb - DH_;
    #pragma unroll
    for (int r = 0; r < 8; r++)
      Vt[((size_t)b * DH_ + d0 + r) * JP_ + j] = (bf16)(acc[r] + bc[cb + r]);
  }
}

// null-kv row (j = 256) + zero the J..JP padding
__global__ void kv_fixup_kernel(const float* __restrict__ null_kv,
                                bf16* __restrict__ Kbuf, bf16* __restrict__ Vt) {
  int b = blockIdx.x, d = threadIdx.x;           // 4 x 64
  Kbuf[((size_t)b * JP_ + M_) * DH_ + d] = (bf16)null_kv[d];
  Vt[((size_t)b * DH_ + d) * JP_ + M_]   = (bf16)null_kv[DH_ + d];
  for (int j = J_; j < JP_; j++) {
    Kbuf[((size_t)b * JP_ + j) * DH_ + d] = (bf16)0.f;
    Vt[((size_t)b * DH_ + d) * JP_ + j]   = (bf16)0.f;
  }
}

// -------------------------------------------------------- flash attention
// One wave handles 16 queries of one (b,h). Computes S^T = K_tile * q^T so each
// lane owns one query column -> softmax stats need only a shfl_xor(16).
// att_bias (604 MB, read-once) is streamed with NT loads + prefetch so L2 keeps
// the reused tensors (q / K / V^T / att) resident.

__global__ void attn_kernel(const bf16* __restrict__ qbuf, const bf16* __restrict__ Kbuf,
                            const bf16* __restrict__ Vt, const float* __restrict__ bias,
                            bf16* __restrict__ att) {
  int lane = threadIdx.x & 31;
  int gw = blockIdx.x * 8 + (threadIdx.x >> 5);  // 4096 waves total
  int ntile = gw & (N_ / 16 - 1);
  int h = (gw >> 7) & (H_ - 1);
  int b = gw >> 10;
  int n0 = ntile << 4;
  int i = lane & 15, half = lane >> 4;

  // q^T as two B operands (d 0..31 and d 32..63); lane col = query i
  bf16x16 qb0, qb1;
  {
    const bf16* p = qbuf + (size_t)(b * N_ + n0 + i) * HD_ + h * DH_ + (half << 4);
    qb0 = *(const bf16x16*)p;
    qb1 = *(const bf16x16*)(p + 32);
  }
  const bf16* Kb = Kbuf + (size_t)b * JP_ * DH_;
  const bf16* Vb = Vt + (size_t)b * DH_ * JP_;
  const float* brow = bias + ((size_t)(b * H_ + h) * N_ + n0 + i) * J_;

  // prefetch the first bias lines for this row
  __builtin_prefetch(brow + (half << 3), 0, 0);

  float m = -3e38f, l = 0.f;
  f32x8 acc[4] = {};                // att^T, 4 d-tiles of 16
  for (int j0 = 0; j0 < JP_; j0 += 32) {
    // prefetch bias two j-blocks ahead (uniform branch)
    if (j0 + 64 < J_) __builtin_prefetch(brow + j0 + 64 + (half << 3), 0, 0);

    f32x8 s0 = {}, s1 = {};
    s0 = wmma_bf16(load_a_op(Kb, DH_, j0,      0,  lane), qb0, s0);
    s0 = wmma_bf16(load_a_op(Kb, DH_, j0,      32, lane), qb1, s0);
    s1 = wmma_bf16(load_a_op(Kb, DH_, j0 + 16, 0,  lane), qb0, s1);
    s1 = wmma_bf16(load_a_op(Kb, DH_, j0 + 16, 32, lane), qb1, s1);

    int jb0 = j0 + (half << 3);     // this lane's j range in S^T tile 0
    int jb1 = jb0 + 16;             // in tile 1
    float p0[8], p1[8];
    float mx = -3e38f;
    #pragma unroll
    for (int r = 0; r < 8; r++) {
      p0[r] = (jb0 + r < J_) ? s0[r] + __builtin_nontemporal_load(brow + jb0 + r)
                             : -3e38f;
      p1[r] = (jb1 + r < J_) ? s1[r] + __builtin_nontemporal_load(brow + jb1 + r)
                             : -3e38f;
      mx = fmaxf(mx, fmaxf(p0[r], p1[r]));
    }
    mx = fmaxf(mx, __shfl_xor(mx, 16));
    float mnew  = fmaxf(m, mx);
    float alpha = __expf(m - mnew);
    float ps = 0.f;
    #pragma unroll
    for (int r = 0; r < 8; r++) {
      p0[r] = __expf(p0[r] - mnew);
      p1[r] = __expf(p1[r] - mnew);
      ps += p0[r] + p1[r];
    }
    ps += __shfl_xor(ps, 16);
    l = l * alpha + ps;
    m = mnew;

    // redistribute P^T into a 32x16 B operand (K=j, col=query)
    bf16x16 pb;
    #pragma unroll
    for (int e = 0; e < 8; e++) {
      float o0 = __shfl_xor(p0[e], 16);
      float o1 = __shfl_xor(p1[e], 16);
      if (half == 0) { pb[e] = (bf16)p0[e]; pb[8 + e] = (bf16)o0; }
      else           { pb[e] = (bf16)o1;    pb[8 + e] = (bf16)p1[e]; }
    }
    // att^T += V^T(d-tile, j-chunk) * P^T, with online-softmax rescale
    #pragma unroll
    for (int t = 0; t < 4; t++) {
      f32x8 a = acc[t];
      #pragma unroll
      for (int r = 0; r < 8; r++) a[r] *= alpha;
      acc[t] = wmma_bf16(load_a_op(Vb + (size_t)t * 16 * JP_, JP_, 0, j0, lane), pb, a);
    }
  }

  float inv = 1.f / l;
  int nrow = b * N_ + n0 + i;
  #pragma unroll
  for (int t = 0; t < 4; t++) {
    bf16x8 o;
    #pragma unroll
    for (int r = 0; r < 8; r++) o[r] = (bf16)(acc[t][r] * inv);
    *(bf16x8*)(att + (size_t)nrow * HD_ + h * DH_ + t * 16 + (half << 3)) = o;
  }
}

// ---------------------------------------------------------------- out proj

__global__ void gemm_out_kernel(const bf16* __restrict__ Wot, const bf16* __restrict__ att,
                                float* __restrict__ y) {
  int lane = threadIdx.x & 31;
  int tile = blockIdx.x * 8 + (threadIdx.x >> 5);
  int c0 = (tile & 63) << 4;   // 1024/16 feature tiles
  int n0 = (tile >> 6) << 4;
  f32x8 acc = {};
  for (int k0 = 0; k0 < HD_; k0 += 32)
    acc = wmma_bf16(load_a_op(Wot, HD_, c0, k0, lane),
                    load_b_op(att, HD_, n0, k0, lane), acc);
  int n = n0 + (lane & 15);
  *(f32x8*)(y + (size_t)n * D_ + c0 + ((lane >> 4) << 3)) = acc;
}

// ---------------------------------------------------------------- launch

extern "C" void kernel_launch(void* const* d_in, const int* in_sizes, int n_in,
                              void* d_out, int out_size, void* d_ws, size_t ws_size,
                              hipStream_t stream) {
  (void)in_sizes; (void)n_in; (void)out_size; (void)ws_size;
  const float* x        = (const float*)d_in[0];
  const float* context  = (const float*)d_in[1];
  const float* att_bias = (const float*)d_in[2];
  const float* ln_g     = (const float*)d_in[3];
  const float* ln_b     = (const float*)d_in[4];
  const float* null_kv  = (const float*)d_in[5];
  const float* Wq       = (const float*)d_in[6];
  const float* Wkv      = (const float*)d_in[7];
  const float* cln_g    = (const float*)d_in[8];
  const float* cln_b    = (const float*)d_in[9];
  const float* Wc       = (const float*)d_in[10];
  const float* bc       = (const float*)d_in[11];
  const float* Wo       = (const float*)d_in[12];
  const float* oln_g    = (const float*)d_in[13];
  const float* oln_b    = (const float*)d_in[14];
  float* out = (float*)d_out;

  char* p = (char*)d_ws;
  auto alloc = [&](size_t bytes) -> char* {
    char* r = p; p += (bytes + 255) & ~(size_t)255; return r;
  };
  bf16*  xn   = (bf16*)alloc((size_t)ROWS_ * D_ * 2);       // 16 MB
  bf16*  cn   = (bf16*)alloc((size_t)CROWS_ * CD_ * 2);     // 1.5 MB
  bf16*  Wqt  = (bf16*)alloc((size_t)HD_ * D_ * 2);         // 1 MB
  bf16*  Wkvt = (bf16*)alloc((size_t)(2 * DH_) * D_ * 2);   // 0.25 MB
  bf16*  Wct  = (bf16*)alloc((size_t)(2 * DH_) * CD_ * 2);  // 0.19 MB
  bf16*  Wot  = (bf16*)alloc((size_t)D_ * HD_ * 2);         // 1 MB
  bf16*  qbuf = (bf16*)alloc((size_t)ROWS_ * HD_ * 2);      // 8 MB
  bf16*  Kbuf = (bf16*)alloc((size_t)B_ * JP_ * DH_ * 2);   // 1.2 MB
  bf16*  Vt   = (bf16*)alloc((size_t)B_ * DH_ * JP_ * 2);   // 1.2 MB
  bf16*  att  = (bf16*)alloc((size_t)ROWS_ * HD_ * 2);      // 8 MB
  float* y    = (float*)alloc((size_t)ROWS_ * D_ * 4);      // 32 MB

  transpose_bf16<<<(D_ * HD_ + TPB - 1) / TPB, TPB, 0, stream>>>(Wq, Wqt, D_, HD_);
  transpose_bf16<<<(D_ * 2 * DH_ + TPB - 1) / TPB, TPB, 0, stream>>>(Wkv, Wkvt, D_, 2 * DH_);
  transpose_bf16<<<(CD_ * 2 * DH_ + TPB - 1) / TPB, TPB, 0, stream>>>(Wc, Wct, CD_, 2 * DH_);
  transpose_bf16<<<(HD_ * D_ + TPB - 1) / TPB, TPB, 0, stream>>>(Wo, Wot, HD_, D_);

  ln_rows_bf16<<<ROWS_, TPB, 0, stream>>>(x, ln_g, ln_b, xn, D_);
  ln_rows_bf16<<<CROWS_, TPB, 0, stream>>>(context, cln_g, cln_b, cn, CD_);

  gemm_q_kernel<<<(ROWS_ / 16) * (HD_ / 16) / 8, TPB, 0, stream>>>(Wqt, xn, qbuf);
  gemm_kv_kernel<<<(ROWS_ / 16) * (2 * DH_ / 16) / 8, TPB, 0, stream>>>(Wkvt, xn, Kbuf, Vt);
  gemm_ctx_kernel<<<(CROWS_ / 16) * (2 * DH_ / 16) / 8, TPB, 0, stream>>>(Wct, cn, bc, Kbuf, Vt);
  kv_fixup_kernel<<<B_, DH_, 0, stream>>>(null_kv, Kbuf, Vt);

  attn_kernel<<<(B_ * H_ * (N_ / 16)) / 8, TPB, 0, stream>>>(qbuf, Kbuf, Vt, att_bias, att);

  gemm_out_kernel<<<(ROWS_ / 16) * (D_ / 16) / 8, TPB, 0, stream>>>(Wot, att, y);
  ln_rows_f32<<<ROWS_, TPB, 0, stream>>>(y, oln_g, oln_b, out, D_);
}
